// CostVolume_14766097564070
// MI455X (gfx1250) — matrix operand order repaired and checked by reading, not compiled, not run on hardware
//
#include <hip/hip_runtime.h>
#include <hip/hip_bf16.h>

// ---------------------------------------------------------------------------
// Cost volume: out[n,u,h,w] = sum_c t[n,c,h,w] * (conv(s)[n,u*C+c,h,w] + b)
// Implicit GEMM (M=8192 px, N=U*C=10368, K=C*9*9=10368), bf16 WMMA, f32 acc,
// epilogue-fused channel contraction. Block tile: 32 pixels x (3u x 128c),
// 8 waves, each wave: 2 M-subtiles x 3 u x 4 K-subchunks = 24 WMMA per (ky,kx).
// Weights pre-transposed once to bf16 [kblock32][uc][32] -> contiguous 32B
// B fragments per lane. A patch staged to LDS in exact WMMA A-fragment layout
// (double buffered, one barrier per (ky,kx)); zero-padding handled by a
// zeroed pad plane + per-lane base-pointer select (no per-element cndmask).
// ---------------------------------------------------------------------------

#define N_DIM 2
#define C_DIM 128
#define H_DIM 64
#define W_DIM 64
#define U_DIM 81
#define KW    9
#define UC    (U_DIM * C_DIM)        // 10368
#define KDIM  (C_DIM * KW * KW)      // 10368

#define WT_BYTES   (214990848ULL)    // 324 * 10368 * 32 * 2
#define SBF_BYTES  (4194304ULL)      // 2 x (2*128*64*64*2); half used
#define SBF_ELEMS  (N_DIM * C_DIM * H_DIM * W_DIM)
#define ZPAD_BYTES (1048576ULL)      // covers 4095 + 15*32768 ushorts < 1 MB

typedef __attribute__((ext_vector_type(16))) __bf16 v16bf;
typedef __attribute__((ext_vector_type(8)))  float  v8f;

__device__ __forceinline__ unsigned short f2bf(float f) {
    unsigned u = __float_as_uint(f);
    unsigned r = u + 0x7FFFu + ((u >> 16) & 1u);   // round-to-nearest-even
    return (unsigned short)(r >> 16);
}

// --- prep 0: zero the pad plane ---------------------------------------------
__global__ __launch_bounds__(256) void cv_zero(unsigned* __restrict__ p, int n) {
    int i = blockIdx.x * 256 + threadIdx.x;
    if (i < n) p[i] = 0u;
}

// --- prep 1: feats_source f32 -> bf16 ---------------------------------------
__global__ __launch_bounds__(256) void cv_cvt_src(const float* __restrict__ src,
                                                  unsigned short* __restrict__ dst,
                                                  int n) {
    int i = blockIdx.x * 256 + threadIdx.x;
    if (i < n) dst[i] = f2bf(src[i]);
}

// --- prep 2: conv_w [uc][c'][ky][kx] f32 -> Wt [kb][uc][j] bf16 -------------
// kb = (ky*9+kx)*4 + c'/32 ; j = c'%32
__global__ __launch_bounds__(256) void cv_prep_wt(const float* __restrict__ w,
                                                  unsigned short* __restrict__ wt) {
    int i = blockIdx.x * 256 + threadIdx.x;        // 107,495,424 elements
    if (i >= KDIM * UC) return;
    int j  = i & 31;
    int t1 = i >> 5;
    int kb = t1 / UC;
    int uc = t1 - kb * UC;
    int cp = ((kb & 3) << 5) | j;                  // c' 0..127
    int kk = kb >> 2;                              // ky*9+kx
    wt[i] = f2bf(w[(size_t)uc * KDIM + (size_t)cp * (KW*KW) + kk]);
}

// --- main: fused implicit-GEMM + contraction --------------------------------
__global__ __launch_bounds__(256) void cv_main(
    const unsigned short* __restrict__ sbf,   // bf16 source feats [n][c][h][w]
    const unsigned short* __restrict__ zpad,  // >=1MB of zeros
    const unsigned short* __restrict__ wt,    // bf16 weights [kb][uc][32]
    const float* __restrict__ tgt,            // f32 target feats [n][c][h][w]
    const float* __restrict__ bias,           // f32 conv bias [UC]
    float* __restrict__ out)                  // f32 [n][U][h][w]
{
    // A fragments: [buf(2)][m2(2)*s(4)][lane(32)][elem(16)] bf16
    __shared__ __align__(32) unsigned short sFrag[2 * 8 * 32 * 16];   // 16 KB
    __shared__ float sPart[8][3][32];                                 // 3 KB

    const int tid  = threadIdx.x;
    const int lane = tid & 31;
    const int wv   = tid >> 5;                 // wave 0..7 -> c slice wv*16..
    const int bx   = blockIdx.x;               // M tile: 256 = 2n * 64h * 2wt
    const int u0   = blockIdx.y * 3;           // 0,3,..,78 ; 81 = 3*27 exact

    const int n   = bx >> 7;
    const int rem = bx & 127;
    const int h   = rem >> 1;
    const int w0  = (rem & 1) << 5;            // 32-pixel tile

    const int khi = lane >> 4;                 // K-half selector for A/B frags
    const int nlo = lane & 15;                 // N (c) within 16
    const int c_local = wv * 16 + nlo;         // 0..127
    const size_t uc0  = (size_t)u0 * C_DIM + c_local;
    const size_t uc1  = uc0 + C_DIM;
    const size_t uc2  = uc0 + 2 * C_DIM;

    // staging decomposition: lanes vary over pixel (coalesced 64B segments)
    const int st_i   = tid & 31;               // pixel 0..31
    const int st_cp0 = tid >> 5;               // c' base 0..7 ; cp = st_cp0+8q
    const int st_m2  = st_i >> 4;
    unsigned lds_dst[16];                      // kk-invariant LDS dests
    #pragma unroll
    for (int q = 0; q < 16; ++q) {
        const int s   = q >> 2;
        const int hia = (q >> 1) & 1;
        const int e   = st_cp0 + ((q & 1) << 3);
        lds_dst[q] = ((((st_m2 << 2) | s) << 5 | (hia << 4) | (st_i & 15)) << 4) | e;
    }
    const unsigned short* sbase =
        sbf + (size_t)(n * C_DIM + st_cp0) * (H_DIM * W_DIM);

    v8f a00 = {}, a01 = {}, a02 = {}, a10 = {}, a11 = {}, a12 = {};

    int kk = 0;
    for (int ky = 0; ky < KW; ++ky) {
        const int hh = h + ky - (KW / 2);
        const int hc = min(max(hh, 0), H_DIM - 1);
        const bool okh = (hh >= 0) & (hh < H_DIM);
        for (int kx = 0; kx < KW; ++kx, ++kk) {
            const int ww  = w0 + st_i + kx - (KW / 2);
            const int wc  = min(max(ww, 0), W_DIM - 1);
            const bool inb = okh & (ww >= 0) & (ww < W_DIM);
            unsigned short* buf = &sFrag[(kk & 1) * 4096];

            // stage: one base-pointer select, then 16 unconditional loads
            const unsigned short* base = inb ? sbase : zpad;
            const unsigned short* gp = base + (hc * W_DIM + wc);
            #pragma unroll
            for (int q = 0; q < 16; ++q)
                buf[lds_dst[q]] = gp[q * 8 * H_DIM * W_DIM];
            __syncthreads();

            // compute: 2 M-subtiles x 3 u x 4 K-subchunks
            #pragma unroll
            for (int s = 0; s < 4; ++s) {
                v16bf A0 = *(const v16bf*)&buf[((s << 5) | lane) << 4];
                v16bf A1 = *(const v16bf*)&buf[(((4 + s) << 5) | lane) << 4];
                const size_t kb = (size_t)kk * 4 + s;
                v16bf B0 = *(const v16bf*)&wt[(kb * UC + uc0) * 32 + khi * 16];
                v16bf B1 = *(const v16bf*)&wt[(kb * UC + uc1) * 32 + khi * 16];
                v16bf B2 = *(const v16bf*)&wt[(kb * UC + uc2) * 32 + khi * 16];
                a00 = __builtin_amdgcn_wmma_f32_16x16x32_bf16(
                    false, A0, false, B0, (short)0, a00, false, false);
                a01 = __builtin_amdgcn_wmma_f32_16x16x32_bf16(
                    false, A0, false, B1, (short)0, a01, false, false);
                a02 = __builtin_amdgcn_wmma_f32_16x16x32_bf16(
                    false, A0, false, B2, (short)0, a02, false, false);
                a10 = __builtin_amdgcn_wmma_f32_16x16x32_bf16(
                    false, A1, false, B0, (short)0, a10, false, false);
                a11 = __builtin_amdgcn_wmma_f32_16x16x32_bf16(
                    false, A1, false, B1, (short)0, a11, false, false);
                a12 = __builtin_amdgcn_wmma_f32_16x16x32_bf16(
                    false, A1, false, B2, (short)0, a12, false, false);
            }
        }
    }

    // Epilogue: (lam + bias) * target, reduce over the 128 c's.
    // Lane holds D[M = khi*8 + r][N = c_local]; pixel = m2*16 + khi*8 + r.
    const float bi0 = bias[uc0];
    const float bi1 = bias[uc1];
    const float bi2 = bias[uc2];
    const float* trow = &tgt[(size_t)(n * C_DIM + c_local) * (H_DIM * W_DIM)
                             + h * W_DIM + w0];
    float p00[8], p01[8], p02[8], p10[8], p11[8], p12[8];
    #pragma unroll
    for (int r = 0; r < 8; ++r) {
        const float t0 = trow[khi * 8 + r];
        const float t1 = trow[16 + khi * 8 + r];
        p00[r] = (a00[r] + bi0) * t0;
        p01[r] = (a01[r] + bi1) * t0;
        p02[r] = (a02[r] + bi2) * t0;
        p10[r] = (a10[r] + bi0) * t1;
        p11[r] = (a11[r] + bi1) * t1;
        p12[r] = (a12[r] + bi2) * t1;
    }
    #pragma unroll
    for (int off = 1; off < 16; off <<= 1) {
        #pragma unroll
        for (int r = 0; r < 8; ++r) {
            p00[r] += __shfl_xor(p00[r], off, 32);
            p01[r] += __shfl_xor(p01[r], off, 32);
            p02[r] += __shfl_xor(p02[r], off, 32);
            p10[r] += __shfl_xor(p10[r], off, 32);
            p11[r] += __shfl_xor(p11[r], off, 32);
            p12[r] += __shfl_xor(p12[r], off, 32);
        }
    }
    __syncthreads();                 // last sFrag reads complete
    if (nlo == 0) {                  // lanes 0 (px 0..7) and 16 (px 8..15)
        #pragma unroll
        for (int r = 0; r < 8; ++r) {
            sPart[wv][0][khi * 8 + r]      = p00[r];
            sPart[wv][1][khi * 8 + r]      = p01[r];
            sPart[wv][2][khi * 8 + r]      = p02[r];
            sPart[wv][0][16 + khi * 8 + r] = p10[r];
            sPart[wv][1][16 + khi * 8 + r] = p11[r];
            sPart[wv][2][16 + khi * 8 + r] = p12[r];
        }
    }
    __syncthreads();
    if (tid < 96) {                  // deterministic fixed-order 8-wave sum
        const int ut  = tid >> 5;    // 0..2
        const int pix = tid & 31;
        float x = 0.f;
        #pragma unroll
        for (int w = 0; w < 8; ++w) x += sPart[w][ut][pix];
        out[((size_t)(n * U_DIM + u0 + ut) * H_DIM + h) * W_DIM + w0 + pix] = x;
    }
}

extern "C" void kernel_launch(void* const* d_in, const int* in_sizes, int n_in,
                              void* d_out, int out_size, void* d_ws, size_t ws_size,
                              hipStream_t stream) {
    const float* tgt = (const float*)d_in[0];   // feats_target
    const float* src = (const float*)d_in[1];   // feats_source
    const float* cw  = (const float*)d_in[2];   // conv_w
    const float* cb  = (const float*)d_in[3];   // conv_b
    float* out = (float*)d_out;

    unsigned short* Wt   = (unsigned short*)d_ws;
    unsigned short* Sbf  = (unsigned short*)((char*)d_ws + WT_BYTES);
    unsigned short* Zpad = (unsigned short*)((char*)d_ws + WT_BYTES + SBF_BYTES);

    cv_zero<<<(int)(ZPAD_BYTES / 4 + 255) / 256, 256, 0, stream>>>(
        (unsigned*)Zpad, (int)(ZPAD_BYTES / 4));
    cv_cvt_src<<<(SBF_ELEMS + 255) / 256, 256, 0, stream>>>(src, Sbf, SBF_ELEMS);
    cv_prep_wt<<<((KDIM * UC) + 255) / 256, 256, 0, stream>>>(cw, Wt);
    cv_main<<<dim3(256, U_DIM / 3), 256, 0, stream>>>(Sbf, Zpad, Wt, tgt, cb, out);
}
